// VQVAE_45775761441268
// MI455X (gfx1250) — matrix operand order repaired and checked by reading, not compiled, not run on hardware
//
#include <hip/hip_runtime.h>

// ---------------------------------------------------------------------------
// MI455X (gfx1250, wave32) VQ-VAE training step.
// Heavy GEMMs: v_wmma_f32_16x16x32_bf16 (fp32 accumulate).
// Data movement: TDM tensor_load_to_lds (k1 x-tile) + async global->LDS (k3),
// both guarded with __has_builtin fallbacks.
// ---------------------------------------------------------------------------

typedef __bf16 v16bf __attribute__((ext_vector_type(16)));
typedef float  v8f   __attribute__((ext_vector_type(8)));
typedef unsigned int u32x4 __attribute__((ext_vector_type(4)));
typedef int          i32x4 __attribute__((ext_vector_type(4)));
typedef int          i32x8 __attribute__((ext_vector_type(8)));

union BF16Frag {
    v16bf          v;
    unsigned short h[16];
    unsigned int   w[8];
    uint4          q[2];
};

static __device__ __forceinline__ unsigned short f2bf(float f) {
    unsigned u = __builtin_bit_cast(unsigned, f);
    unsigned r = (u + 0x7FFFu + ((u >> 16) & 1u)) >> 16;
    return (unsigned short)r;
}

static __device__ __forceinline__ v8f vzero8() {
    v8f z;
#pragma unroll
    for (int i = 0; i < 8; ++i) z[i] = 0.0f;
    return z;
}

#define NT   65536
#define DIM  256
#define KC   512

// ---- gfx1250 data-mover paths (guarded) -----------------------------------
#define AS1 __attribute__((address_space(1)))
#define AS3 __attribute__((address_space(3)))

#if defined(__gfx1250__) && __has_builtin(__builtin_amdgcn_tensor_load_to_lds)
#define HAVE_TDM 1
#endif
#if defined(__gfx1250__) && __has_builtin(__builtin_amdgcn_global_load_async_to_lds_b128)
#define HAVE_ASYNC 1
static __device__ __forceinline__ void async_copy16(const void* g, void* l) {
    __builtin_amdgcn_global_load_async_to_lds_b128(
        (AS1 i32x4*)g, (AS3 i32x4*)l, 0, 0);
}
static __device__ __forceinline__ void async_wait0() {
#if __has_builtin(__builtin_amdgcn_s_wait_asynccnt)
    __builtin_amdgcn_s_wait_asynccnt(0);
#else
    asm volatile("s_wait_asynccnt 0x0" ::: "memory");
#endif
}
#endif

static __device__ __forceinline__ void tensor_wait0() {
#if __has_builtin(__builtin_amdgcn_s_wait_tensorcnt)
    __builtin_amdgcn_s_wait_tensorcnt(0);
#else
    asm volatile("s_wait_tensorcnt 0x0" ::: "memory");
#endif
}

// ---------------------------------------------------------------------------
// k0: E -> bf16 (row major + transposed), ||e||^2, zero accumulators.
// ---------------------------------------------------------------------------
__global__ __launch_bounds__(256) void k0_prep(const float* __restrict__ E,
                                               unsigned short* __restrict__ Ebf,
                                               unsigned short* __restrict__ ETbf,
                                               float* __restrict__ esq,
                                               float* __restrict__ colsum,
                                               float* __restrict__ dw,
                                               float* __restrict__ mseacc) {
    const int k = blockIdx.x;
    const int d = threadIdx.x;
    float v = E[k * DIM + d];
    unsigned short b = f2bf(v);
    Ebf[k * DIM + d] = b;
    ETbf[d * KC + k] = b;        // transposed copy for the enc@E GEMM
    dw[k * DIM + d]  = 0.0f;     // k3 accumulates atomically
    __shared__ float red[256];
    red[d] = v * v;
    __syncthreads();
#pragma unroll
    for (int s = 128; s > 0; s >>= 1) {
        if (d < s) red[d] += red[d + s];
        __syncthreads();
    }
    if (d == 0) esq[k] = red[0];
    if (k == 0) {
        colsum[d]       = 0.0f;
        colsum[d + 256] = 0.0f;
        if (d == 0) mseacc[0] = 0.0f;
    }
}

// ---------------------------------------------------------------------------
// k1: scores = 2*x@E^T - ||e||^2, fused softmax, enc(bf16) out, column sums.
// grid: N/32 blocks x 64 threads (2 waves, 16 rows each)
// LDS: 32x512 f32 (64 KB); x staged by TDM with HW padding (stride 260 f32),
// then reused as the score/enc matrix (stride 512).
// ---------------------------------------------------------------------------
#define XS_STRIDE 260

__global__ __launch_bounds__(64) void k1_scores(const float* __restrict__ x,
                                                const unsigned short* __restrict__ Ebf,
                                                const float* __restrict__ esq,
                                                unsigned short* __restrict__ encbf,
                                                float* __restrict__ colsum) {
    __shared__ __align__(16) float smem[32 * 512];
    const int tid  = threadIdx.x;
    const int wave = tid >> 5;
    const int lane = tid & 31;
    const int row0 = blockIdx.x * 32;
    const int lm   = lane & 15;
    const int hi   = lane >> 4;

    // ---- stage 32x256 x-tile into LDS (padded stride 260) ----
#if HAVE_TDM
    if (wave == 0) {
        // Tensor DMA descriptor (ISA ch.8): 4B elements, tile 256x32 of a
        // row-major [65536 x 256] tensor starting at row0; pad 4 DW per 256 DW
        // so the LDS tile lands with stride 260 floats (bank-conflict free).
        unsigned long long ga =
            (unsigned long long)(const void*)(x + (size_t)row0 * DIM);
        u32x4 g0;
        g0[0] = 1u;                                         // count=1 (user D#)
        g0[1] = 0u;                                         // lds_addr = 0 (smem base)
        g0[2] = (unsigned)(ga & 0xffffffffu);               // global_addr[31:0]
        g0[3] = (unsigned)((ga >> 32) & 0x01FFFFFFu) | (2u << 30); // addr[56:32], type=2
        i32x8 g1;
        g1[0] = (int)(0x00020000u          // data_size = 2 (4 bytes)
                      | (1u << 20)         // pad_enable
                      | (7u << 22)         // pad_interval: 256 DWORDs
                      | (3u << 25));       // pad_amount:   4 DWORDs
        g1[1] = (int)(256u << 16);         // tensor_dim0 = 256 (low 16)
        g1[2] = 0;                         // dim0 hi=0 | dim1 low16 (65536 -> 0)
        g1[3] = (int)(1u | (256u << 16));  // dim1 hi=1 | tile_dim0 = 256
        g1[4] = 32;                        // tile_dim1 = 32 | tile_dim2 = 0
        g1[5] = 256;                       // tensor_dim0_stride = 256
        g1[6] = 0;
        g1[7] = 0;
        i32x4 gz = {0, 0, 0, 0};
#if __clang_major__ >= 23
        i32x8 gz8 = {0, 0, 0, 0, 0, 0, 0, 0};
        __builtin_amdgcn_tensor_load_to_lds(g0, g1, gz, gz, gz8, 0);
#else
        __builtin_amdgcn_tensor_load_to_lds(g0, g1, gz, gz, 0);
#endif
        tensor_wait0();
    }
#else
    for (int idx = tid; idx < 32 * 64; idx += 64) {
        int r = idx >> 6, c4 = idx & 63;
        float4 v = ((const float4*)(x + (size_t)(row0 + r) * DIM))[c4];
        float* dst = &smem[r * XS_STRIDE + c4 * 4];
        dst[0] = v.x; dst[1] = v.y; dst[2] = v.z; dst[3] = v.w;
    }
#endif
    __syncthreads();

    // ---- build 8 A fragments (16 rows x 256 dims, bf16) per ISA A-layout ----
    const int rloc = wave * 16 + lm;
    BF16Frag A[8];
#pragma unroll
    for (int j = 0; j < 8; ++j) {
#pragma unroll
        for (int e = 0; e < 16; ++e) {
            int dd = 32 * j + (e & 7) + 8 * (((e >> 3) << 1) + hi);
            A[j].h[e] = f2bf(smem[rloc * XS_STRIDE + dd]);
        }
    }
    __syncthreads();   // everyone done reading x before scores overwrite LDS

    // ---- 32 code tiles of 16: scores via K-chained WMMA ----
    float* myScores = &smem[(wave * 16) * 512];
    for (int c = 0; c < 32; ++c) {
        int n = c * 16 + lm;
        float eq = esq[n];
        const uint4* erow = (const uint4*)(Ebf + (size_t)n * DIM);
        v8f acc = vzero8();
#pragma unroll
        for (int j = 0; j < 8; ++j) {
            BF16Frag B;
            B.q[0] = erow[4 * j + hi];
            B.q[1] = erow[4 * j + 2 + hi];
            acc = __builtin_amdgcn_wmma_f32_16x16x32_bf16(
                false, A[j].v, false, B.v, (short)0, acc, false, false);
        }
#pragma unroll
        for (int v = 0; v < 8; ++v) {               // C layout: M = v + 8*hi
            int r = v + 8 * hi;
            myScores[r * 512 + n] = 2.0f * acc[v] - eq;
        }
    }
    __syncthreads();

    // ---- fused softmax over K=512 (||x||^2 cancels); 2 threads per row ----
    {
        int r = tid >> 1, half = tid & 1;
        float* srow = &smem[r * 512 + half * 256];
        float m = -3.4e38f;
        for (int i = 0; i < 256; ++i) m = fmaxf(m, srow[i]);
        m = fmaxf(m, __shfl_xor(m, 1));
        float sum = 0.0f;
        for (int i = 0; i < 256; ++i) { float e = __expf(srow[i] - m); srow[i] = e; sum += e; }
        sum += __shfl_xor(sum, 1);
        float inv = 1.0f / sum;
        unsigned int* dst =
            (unsigned int*)(encbf + (size_t)(row0 + r) * KC + half * 256);
        for (int i = 0; i < 128; ++i) {
            float v0 = srow[2 * i] * inv, v1 = srow[2 * i + 1] * inv;
            srow[2 * i] = v0; srow[2 * i + 1] = v1;      // keep for colsum
            dst[i] = (unsigned)f2bf(v0) | ((unsigned)f2bf(v1) << 16);
        }
    }
    __syncthreads();

    // ---- per-column partial sums -> global atomics ----
    for (int c = tid; c < 512; c += 64) {
        float s = 0.0f;
        for (int r = 0; r < 32; ++r) s += smem[r * 512 + c];
        atomicAdd(&colsum[c], s);
    }
}

// ---------------------------------------------------------------------------
// k2: quantized = enc @ E ; fused ST output (x + (q-x)) and MSE reduction.
// grid: N/64 blocks x 128 threads (4 waves, 16 rows each, full 256 cols)
// ---------------------------------------------------------------------------
__global__ __launch_bounds__(128) void k2_quantize(const unsigned short* __restrict__ encbf,
                                                   const unsigned short* __restrict__ ETbf,
                                                   const float* __restrict__ x,
                                                   float* __restrict__ out,
                                                   float* __restrict__ mseacc) {
    const int tid  = threadIdx.x;
    const int wave = tid >> 5;
    const int lane = tid & 31;
    const int lm   = lane & 15;
    const int hi   = lane >> 4;
    const int r0   = blockIdx.x * 64 + wave * 16;

    v8f acc[16];
#pragma unroll
    for (int t = 0; t < 16; ++t) acc[t] = vzero8();

    const uint4* arow = (const uint4*)(encbf + (size_t)(r0 + lm) * KC);
    for (int j = 0; j < 16; ++j) {                    // K = 512 codes / 32
        BF16Frag A;
        A.q[0] = arow[4 * j + hi];
        A.q[1] = arow[4 * j + 2 + hi];
#pragma unroll
        for (int t = 0; t < 16; ++t) {
            BF16Frag B;
            const uint4* brow = (const uint4*)(ETbf + (size_t)(16 * t + lm) * KC);
            B.q[0] = brow[4 * j + hi];
            B.q[1] = brow[4 * j + 2 + hi];
            acc[t] = __builtin_amdgcn_wmma_f32_16x16x32_bf16(
                false, A.v, false, B.v, (short)0, acc[t], false, false);
        }
    }

    float mse = 0.0f;
#pragma unroll
    for (int t = 0; t < 16; ++t) {
#pragma unroll
        for (int v = 0; v < 8; ++v) {
            int row = r0 + v + 8 * hi;
            int d   = 16 * t + lm;
            float q  = acc[t][v];
            float xv = x[(size_t)row * DIM + d];
            out[(size_t)row * DIM + d] = xv + (q - xv);  // straight-through
            float diff = q - xv;
            mse += diff * diff;
        }
    }
#pragma unroll
    for (int o = 16; o > 0; o >>= 1) mse += __shfl_xor(mse, o);
    if (lane == 0) atomicAdd(mseacc, mse);
}

// ---------------------------------------------------------------------------
// k3: dw = enc^T @ x, split-K over row blocks, atomic f32 accumulate.
// grid: 512 blocks = 8 code-slabs x 64 row-splits; 128 threads (4 waves).
// enc slab staged via async global->LDS copies (ASYNCcnt); x staged with a
// transposing f32->bf16 conversion so B fragments are contiguous b128 reads.
// ---------------------------------------------------------------------------
__global__ __launch_bounds__(128) void k3_dw(const unsigned short* __restrict__ encbf,
                                             const float* __restrict__ x,
                                             float* __restrict__ dw) {
    __shared__ __align__(16) unsigned short encS[32 * 72];   // [n][code], pad 8
    __shared__ __align__(16) unsigned short xT[256 * 40];    // [dim][n],  pad 8
    const int tid   = threadIdx.x;
    const int wave  = tid >> 5;
    const int lane  = tid & 31;
    const int lm    = lane & 15;
    const int hi    = lane >> 4;
    const int kslab = blockIdx.x & 7;               // 64 codes
    const int nb0   = (blockIdx.x >> 3) * 1024;     // 1024 rows per split
    const int c0    = kslab * 64 + wave * 16;

    v8f acc[16];
#pragma unroll
    for (int t = 0; t < 16; ++t) acc[t] = vzero8();

    for (int it = 0; it < 32; ++it) {
        const int nb = nb0 + it * 32;
        {   // stage: thread -> row n = tid/4, quarter seg = tid%4
            int n = tid >> 2, seg = tid & 3;
            const unsigned short* esrc =
                encbf + (size_t)(nb + n) * KC + kslab * 64 + seg * 16;
#if HAVE_ASYNC
            async_copy16(esrc,     &encS[n * 72 + seg * 16]);
            async_copy16(esrc + 8, &encS[n * 72 + seg * 16 + 8]);
#else
            uint4* ed = (uint4*)&encS[n * 72 + seg * 16];
            const uint4* es4 = (const uint4*)esrc;
            ed[0] = es4[0];
            ed[1] = es4[1];
#endif
            const float4* xr =
                (const float4*)(x + (size_t)(nb + n) * DIM + seg * 64);
#pragma unroll
            for (int i = 0; i < 16; ++i) {
                float4 v = xr[i];
                int d = seg * 64 + i * 4;
                xT[(d + 0) * 40 + n] = f2bf(v.x);
                xT[(d + 1) * 40 + n] = f2bf(v.y);
                xT[(d + 2) * 40 + n] = f2bf(v.z);
                xT[(d + 3) * 40 + n] = f2bf(v.w);
            }
#if HAVE_ASYNC
            async_wait0();
#endif
        }
        __syncthreads();

        BF16Frag A;                                  // enc^T: 16 codes x 32 n
#pragma unroll
        for (int e = 0; e < 16; ++e) {
            int nn = (e & 7) + 8 * (((e >> 3) << 1) + hi);
            A.h[e] = encS[nn * 72 + wave * 16 + lm];
        }
#pragma unroll
        for (int t = 0; t < 16; ++t) {
            BF16Frag B;                              // x: 32 n x 16 dims
            const uint4* br = (const uint4*)(&xT[(16 * t + lm) * 40]);
            B.q[0] = br[hi];                         // n = 8*hi .. +7
            B.q[1] = br[2 + hi];
            acc[t] = __builtin_amdgcn_wmma_f32_16x16x32_bf16(
                false, A.v, false, B.v, (short)0, acc[t], false, false);
        }
        __syncthreads();
    }

#pragma unroll
    for (int t = 0; t < 16; ++t) {
#pragma unroll
        for (int v = 0; v < 8; ++v) {
            int code = c0 + v + 8 * hi;
            int d    = 16 * t + lm;
            atomicAdd(&dw[code * DIM + d], acc[t][v]);
        }
    }
}

// ---------------------------------------------------------------------------
// k4: finalize losses. One block, 512 threads (one per code).
// ---------------------------------------------------------------------------
__global__ __launch_bounds__(512) void k4_finalize(const float* __restrict__ ecs,
                                                   const float* __restrict__ ema_w,
                                                   const float* __restrict__ usage,
                                                   const float* __restrict__ colsum,
                                                   const float* __restrict__ dw,
                                                   const float* __restrict__ mseacc,
                                                   float* __restrict__ out_scalars) {
    __shared__ float red[512];
    const int t = threadIdx.x;

    auto blockSum = [&](float v) -> float {
        red[t] = v;
        __syncthreads();
        for (int s = 256; s > 0; s >>= 1) {
            if (t < s) red[t] += red[t + s];
            __syncthreads();
        }
        float r = red[0];
        __syncthreads();
        return r;
    };

    float cs_col = colsum[t];
    float necs = 0.99f * ecs[t]   + 0.01f * cs_col;   // DECAY = 0.99
    float nuc  = 0.99f * usage[t] + 0.01f * cs_col;
    float avg  = cs_col / (float)NT;

    float n    = blockSum(necs);
    float usum = blockSum(nuc);
    float ent  = blockSum(-avg * __logf(avg + 1e-10f));
    float up   = nuc / (usum + 1e-5f);
    float div  = blockSum(-up * __logf(up + 1e-10f));

    float cluster = (necs + 1e-5f) / (n + (float)KC * 1e-5f) * n;
    float inv = 1.0f / cluster;
    float regp = 0.0f;
    const float* wrow = ema_w + t * DIM;
    const float* drow = dw    + t * DIM;
    for (int d = 0; d < DIM; ++d) {
        float ne = (0.99f * wrow[d] + 0.01f * drow[d]) * inv;  // new embedding
        regp += ne * ne;
    }
    float reg = blockSum(regp);

    if (t == 0) {
        float mse  = mseacc[0] / ((float)NT * (float)DIM);
        float loss = 1.25f * mse + reg + 0.8f * (ent + div);   // q + BETA*e
        out_scalars[0] = loss;
        out_scalars[1] = __expf(ent);                          // perplexity
    }
}

// ---------------------------------------------------------------------------
// Host launcher. ws layout (bytes):
//   0        Ebf   512*256*2
//   262144   ETbf  256*512*2
//   524288   esq   512*4
//   526336   colsum 512*4
//   528384   dw    512*256*4
//   1052672  mseacc 4
//   2097152  enc(bf16) 65536*512*2   (total ~66 MB)
// ---------------------------------------------------------------------------
extern "C" void kernel_launch(void* const* d_in, const int* in_sizes, int n_in,
                              void* d_out, int out_size, void* d_ws, size_t ws_size,
                              hipStream_t stream) {
    const float* x     = (const float*)d_in[0];
    const float* E     = (const float*)d_in[1];
    const float* ecs   = (const float*)d_in[2];
    const float* ema_w = (const float*)d_in[3];
    const float* usage = (const float*)d_in[4];
    float* out = (float*)d_out;

    char* ws = (char*)d_ws;
    unsigned short* Ebf   = (unsigned short*)(ws + 0);
    unsigned short* ETbf  = (unsigned short*)(ws + 262144);
    float*          esq   = (float*)(ws + 524288);
    float*          csum  = (float*)(ws + 526336);
    float*          dw    = (float*)(ws + 528384);
    float*          msea  = (float*)(ws + 1052672);
    unsigned short* encbf = (unsigned short*)(ws + 2097152);

    k0_prep    <<<KC,        256, 0, stream>>>(E, Ebf, ETbf, esq, csum, dw, msea);
    k1_scores  <<<NT / 32,    64, 0, stream>>>(x, Ebf, esq, encbf, csum);
    k2_quantize<<<NT / 64,   128, 0, stream>>>(encbf, ETbf, x, out, msea);
    k3_dw      <<<512,       128, 0, stream>>>(encbf, x, dw);
    k4_finalize<<<1,         512, 0, stream>>>(ecs, ema_w, usage, csum, dw, msea,
                                               out + (size_t)NT * DIM);
}
